// CAN_41884521071035
// MI455X (gfx1250) — compile-verified, matrix operand order checked by reading, not run on hardware
//
#include <hip/hip_runtime.h>
#include <hip/hip_bf16.h>
#include <stdint.h>

// MAC-cell recurrence for MI455X (gfx1250, wave32).
// All GEMMs run on v_wmma_f32_16x16x32_bf16 with fp32 accumulation.
// The dominant ReadUnit GEMM is algebraically collapsed (matmul commuted with
// the per-batch contraction vector), turning 13.2 G-MAC/step into ~93 M-MAC/step
// plus one L2-resident streaming pass over KB2/KBp per step.
// GEMM uses a 32x64 wave tile (8 WMMAs / K-step) for ~10.7 flops/byte from L2.

#define BATCH 128
#define DIM   512
#define SLOT  30
#define HWP   196          // 14*14
#define TSTEPS 12          // reference T (steps input is constant 12)
#define NHIST (TSTEPS + 1)

typedef __attribute__((ext_vector_type(16))) __bf16 v16bf;
typedef __attribute__((ext_vector_type(8)))  float  v8f;

struct U4x2 { uint4 lo; uint4 hi; };

__device__ __forceinline__ v16bf load_frag16(const __bf16* p0, const __bf16* p1) {
  U4x2 u;
  u.lo = *(const uint4*)p0;
  u.hi = *(const uint4*)p1;
  return __builtin_bit_cast(v16bf, u);
}

// ---------------------------------------------------------------------------
// Generic bf16 WMMA GEMM:  C(MxN) = A(MxK,row-major,bf16) @ Wp(packed) + bias
// Wp is pre-packed into the CDNA5 16x16x32 B-fragment lane layout so every
// lane reads 32 contiguous bytes per K-step. One wave computes a 32x64 tile
// (2 M-subtiles x 4 N-subtiles -> 8 accumulators); A-frags are reused across
// 4 N-subtiles and B-frags across 2 M-subtiles.
// Block = 128 threads (4 waves along M -> 128x64 block tile).
// grid = (N/64, M/128). Requires M%128==0, N%64==0.
// Optionally writes fp32 (ldc) and/or bf16 (ldcb) outputs.
// ---------------------------------------------------------------------------
__global__ void __launch_bounds__(128)
gemm_bf16_wmma(const __bf16* __restrict__ A, int lda,
               const __bf16* __restrict__ Wp,
               const float* __restrict__ bias,
               float* __restrict__ Cf, int ldc,
               __bf16* __restrict__ Cb, int ldcb,
               int N, int K)
{
  const int lane = threadIdx.x & 31;
  const int wave = threadIdx.x >> 5;
  const int ntg  = blockIdx.x << 2;                    // first of 4 n-tiles
  const int m0   = (blockIdx.y << 7) + (wave << 5);    // wave rows [m0, m0+32)
  const int rowA = m0 + (lane & 15);
  const int kA   = (lane < 16) ? 0 : 8;                // 16-bit A-frag K base
  const int ntiles = N >> 4;
  const int ktiles = K >> 5;

  v8f acc[2][4];
#pragma unroll
  for (int mi = 0; mi < 2; ++mi)
#pragma unroll
    for (int nj = 0; nj < 4; ++nj)
      acc[mi][nj] = (v8f){0.f, 0.f, 0.f, 0.f, 0.f, 0.f, 0.f, 0.f};

  for (int kt = 0; kt < ktiles; ++kt) {
    const __bf16* ap0 = A + (size_t)rowA * lda + (kt << 5) + kA;
    const __bf16* ap1 = ap0 + (size_t)16 * lda;
    const v16bf a0 = load_frag16(ap0, ap0 + 16);       // K runs [b..b+7],[b+16..b+23]
    const v16bf a1 = load_frag16(ap1, ap1 + 16);
#pragma unroll
    for (int nj = 0; nj < 4; ++nj) {
      const __bf16* bp = Wp + (((size_t)(kt * ntiles + ntg + nj)) * 32 + lane) * 16;
      const v16bf bf = load_frag16(bp, bp + 8);        // contiguous packed fragment
      acc[0][nj] = __builtin_amdgcn_wmma_f32_16x16x32_bf16(false, a0, false, bf,
                                                           (short)0, acc[0][nj], false, false);
      acc[1][nj] = __builtin_amdgcn_wmma_f32_16x16x32_bf16(false, a1, false, bf,
                                                           (short)0, acc[1][nj], false, false);
    }
  }

  const int cb   = lane & 15;
  const int rsel = (lane < 16) ? 0 : 8;                // f32 C/D VGPR layout
#pragma unroll
  for (int mi = 0; mi < 2; ++mi) {
#pragma unroll
    for (int nj = 0; nj < 4; ++nj) {
      const int col = ((ntg + nj) << 4) + cb;
      const float bv = bias ? bias[col] : 0.0f;
      const int rbase = m0 + (mi << 4) + rsel;
#pragma unroll
      for (int e = 0; e < 8; ++e) {
        const float v = acc[mi][nj][e] + bv;
        const size_t r = (size_t)(rbase + e);
        if (Cf) Cf[r * (size_t)ldc  + col] = v;
        if (Cb) Cb[r * (size_t)ldcb + col] = (__bf16)v;
      }
    }
  }
}

// Repack a fp32 KxN weight into the WMMA B-fragment lane layout (bf16).
__global__ void pack_weight(const float* __restrict__ W, __bf16* __restrict__ Wp,
                            int K, int N)
{
  const int total = (K >> 5) * (N >> 4) * 256;   // 32 lanes * 8 dwords per tile
  const int idx = blockIdx.x * blockDim.x + threadIdx.x;
  if (idx >= total) return;
  const int j    = idx & 7;
  const int l    = (idx >> 3) & 31;
  const int tile = idx >> 8;
  const int ntiles = N >> 4;
  const int nt = tile % ntiles;
  const int kt = tile / ntiles;
  const int n  = nt * 16 + (l & 15);
  const int kb = ((l < 16) ? 0 : 8) + ((j >= 4) ? 16 : 0);
  const int k0 = kt * 32 + kb + 2 * (j & 3);
  __bf16* dst = Wp + ((size_t)tile * 32 + l) * 16 + 2 * j;
  dst[0] = (__bf16)W[(size_t)k0 * N + n];
  dst[1] = (__bf16)W[(size_t)(k0 + 1) * N + n];
}

// Build packed V (DIM x BATCH), V[d][b] = c_i[b,d] * wra[d], directly in B-frag layout.
__global__ void pack_V(const float* __restrict__ c_cur, const float* __restrict__ wra,
                       __bf16* __restrict__ Vp)
{
  const int total = (DIM >> 5) * (BATCH >> 4) * 256;   // 32768
  const int idx = blockIdx.x * blockDim.x + threadIdx.x;
  if (idx >= total) return;
  const int j    = idx & 7;
  const int l    = (idx >> 3) & 31;
  const int tile = idx >> 8;
  const int ntiles = BATCH >> 4;                       // 8
  const int nt = tile % ntiles;
  const int kt = tile / ntiles;
  const int b  = nt * 16 + (l & 15);
  const int kb = ((l < 16) ? 0 : 8) + ((j >= 4) ? 16 : 0);
  const int k0 = kt * 32 + kb + 2 * (j & 3);
  __bf16* dst = Vp + ((size_t)tile * 32 + l) * 16 + 2 * j;
  dst[0] = (__bf16)(c_cur[(size_t)b * DIM + k0]     * wra[k0]);
  dst[1] = (__bf16)(c_cur[(size_t)b * DIM + k0 + 1] * wra[k0 + 1]);
}

// KB (B,D,14,14) -> KBp bf16 (B,196,D) via LDS-tiled transpose (coalesced both ways).
__global__ void transpose_kb(const float* __restrict__ KB, __bf16* __restrict__ KBp)
{
  __shared__ float tile[32][33];
  const int b  = blockIdx.z;
  const int p0 = blockIdx.x * 32;
  const int d0 = blockIdx.y * 32;
  const float* src = KB + (size_t)b * DIM * HWP;
  for (int r = threadIdx.y; r < 32; r += 8) {
    const int d = d0 + r, p = p0 + threadIdx.x;
    tile[r][threadIdx.x] = (p < HWP) ? src[(size_t)d * HWP + p] : 0.f;
  }
  __syncthreads();
  __bf16* dst = KBp + (size_t)b * HWP * DIM;
  for (int r = threadIdx.y; r < 32; r += 8) {
    const int p = p0 + r, d = d0 + threadIdx.x;
    if (p < HWP) dst[(size_t)p * DIM + d] = (__bf16)tile[threadIdx.x][r];
  }
}

__global__ void zero_f32(float* p, int n) {
  int i = blockIdx.x * blockDim.x + threadIdx.x;
  if (i < n) p[i] = 0.f;
}
__global__ void copy_f32(float* __restrict__ dst, const float* __restrict__ src, int n) {
  int i = blockIdx.x * blockDim.x + threadIdx.x;
  if (i < n) dst[i] = src[i];
}
// dst[r*ld + c] = (bf16) src[r*cols + c]
__global__ void cvt_bf16(const float* __restrict__ src, __bf16* __restrict__ dst,
                         int rows, int cols, int ld) {
  int i = blockIdx.x * blockDim.x + threadIdx.x;
  if (i >= rows * cols) return;
  int r = i / cols, c = i % cols;
  dst[(size_t)r * ld + c] = (__bf16)src[i];
}

// ---------------------------------------------------------------------------
// ControlUnit attention: clog[b,s]=sum_d cq[b,d]*wca[d]*cws[b,d,s]+bca; softmax;
// c_i[b,d]=sum_s attn[s]*cws[b,d,s]. One block per b.
// ---------------------------------------------------------------------------
__global__ void __launch_bounds__(256)
control_attn(const float* __restrict__ cq, const float* __restrict__ cws,
             const float* __restrict__ wca, const float* __restrict__ bca,
             float* __restrict__ c_out, __bf16* __restrict__ catC)
{
  const int b = blockIdx.x, t = threadIdx.x;
  __shared__ float sl[SLOT];
  if (t < SLOT) sl[t] = 0.f;
  __syncthreads();

  float part[SLOT];
#pragma unroll
  for (int s = 0; s < SLOT; ++s) part[s] = 0.f;
  for (int d = t; d < DIM; d += 256) {
    const float w = cq[(size_t)b * DIM + d] * wca[d];
    const float* cw = cws + ((size_t)b * DIM + d) * SLOT;
#pragma unroll
    for (int s = 0; s < SLOT; ++s) part[s] += w * cw[s];
  }
#pragma unroll
  for (int s = 0; s < SLOT; ++s) atomicAdd(&sl[s], part[s]);
  __syncthreads();

  if (t == 0) {
    const float bc = *bca;
    float mx = -1e30f;
    for (int s = 0; s < SLOT; ++s) { sl[s] += bc; mx = fmaxf(mx, sl[s]); }
    float sum = 0.f;
    for (int s = 0; s < SLOT; ++s) { sl[s] = __expf(sl[s] - mx); sum += sl[s]; }
    const float inv = 1.f / sum;
    for (int s = 0; s < SLOT; ++s) sl[s] *= inv;
  }
  __syncthreads();

  for (int d = t; d < DIM; d += 256) {
    const float* cw = cws + ((size_t)b * DIM + d) * SLOT;
    float acc = 0.f;
#pragma unroll
    for (int s = 0; s < SLOT; ++s) acc += sl[s] * cw[s];
    c_out[(size_t)b * DIM + d] = acc;
    catC[(size_t)b * 1024 + DIM + d] = (__bf16)acc;   // c_i bf16 (second half of [q_i|c])
  }
}

// ---------------------------------------------------------------------------
// Fused ReadUnit: rlog[b,p] = KB2[b,p,:].(mp[b]*u1_b) + KBp[b,p,:].u2_b + const_b
// softmax over p, then m_new[b,:] = sum_p attn[p]*KBp[b,p,:].
// One block (256 thr = 8 waves) per b; single L2-resident pass over KB2/KBp.
// ---------------------------------------------------------------------------
__global__ void __launch_bounds__(256)
read_unit(const __bf16* __restrict__ KBp, const __bf16* __restrict__ KB2,
          const float* __restrict__ mp, const float* __restrict__ U,
          const float* __restrict__ c_cur, const float* __restrict__ wra,
          const float* __restrict__ brm, const float* __restrict__ bra,
          __bf16* __restrict__ catM)
{
  const int b = blockIdx.x, t = threadIdx.x;
  const int lane = t & 31, wave = t >> 5;
  __shared__ float a1[DIM], u2[DIM];
  __shared__ float rl[224];
  __shared__ float wred[8];
  __shared__ float rc, smax, sinv;
  if (t == 0) rc = 0.f;
  __syncthreads();

  float cpart = 0.f;
  for (int d = t; d < DIM; d += 256) {
    const float u1v = U[(size_t)d * BATCH + b];
    a1[d] = mp[(size_t)b * DIM + d] * u1v;
    u2[d] = U[(size_t)(DIM + d) * BATCH + b];
    cpart += brm[d] * c_cur[(size_t)b * DIM + d] * wra[d];
  }
  atomicAdd(&rc, cpart);
  __syncthreads();

  const __bf16* kb2b = KB2 + (size_t)b * HWP * DIM;
  const __bf16* kbpb = KBp + (size_t)b * HWP * DIM;
  const float brav = *bra;
  for (int p = wave; p < HWP; p += 8) {
    const __bf16* r2 = kb2b + (size_t)p * DIM;
    const __bf16* rp = kbpb + (size_t)p * DIM;
    __builtin_prefetch(r2 + 8 * DIM, 0, 1);
    __builtin_prefetch(rp + 8 * DIM, 0, 1);
    float s = 0.f;
    for (int d = lane; d < DIM; d += 32)
      s += (float)r2[d] * a1[d] + (float)rp[d] * u2[d];
#pragma unroll
    for (int off = 16; off; off >>= 1) s += __shfl_down(s, off, 32);
    if (lane == 0) rl[p] = s + rc + brav;
  }
  __syncthreads();

  // softmax over 196
  float v = (t < HWP) ? rl[t] : -1e30f;
  float m = v;
#pragma unroll
  for (int off = 16; off; off >>= 1) m = fmaxf(m, __shfl_down(m, off, 32));
  if (lane == 0) wred[wave] = m;
  __syncthreads();
  if (t == 0) { float mm = wred[0]; for (int i = 1; i < 8; ++i) mm = fmaxf(mm, wred[i]); smax = mm; }
  __syncthreads();
  float e = (t < HWP) ? __expf(rl[t] - smax) : 0.f;
  if (t < HWP) rl[t] = e;
  float s2 = e;
#pragma unroll
  for (int off = 16; off; off >>= 1) s2 += __shfl_down(s2, off, 32);
  if (lane == 0) wred[wave] = s2;
  __syncthreads();
  if (t == 0) { float ss = 0.f; for (int i = 0; i < 8; ++i) ss += wred[i]; sinv = 1.f / ss; }
  __syncthreads();
  if (t < HWP) rl[t] *= sinv;
  __syncthreads();

  for (int d = t; d < DIM; d += 256) {
    float acc = 0.f;
    for (int p = 0; p < HWP; ++p) acc += rl[p] * (float)kbpb[(size_t)p * DIM + d];
    catM[(size_t)b * 1024 + d] = (__bf16)acc;        // m_new bf16 (first half of [m_new|m_prev])
  }
}

// ---------------------------------------------------------------------------
// WriteUnit self-attention over history. One block per b.
// ---------------------------------------------------------------------------
__global__ void __launch_bounds__(256)
write_selfattn(const float* __restrict__ c_cur, const float* __restrict__ c_hist,
               const float* __restrict__ m_hist, const float* __restrict__ wwa,
               const float* __restrict__ bwa, int step, __bf16* __restrict__ catA)
{
  const int b = blockIdx.x, t = threadIdx.x;
  __shared__ float sl[NHIST];
  if (t < NHIST) sl[t] = 0.f;
  __syncthreads();

  float part[NHIST];
#pragma unroll
  for (int i = 0; i < NHIST; ++i) part[i] = 0.f;
  for (int d = t; d < DIM; d += 256) {
    const float w = c_cur[(size_t)b * DIM + d] * wwa[d];
#pragma unroll
    for (int i = 0; i < NHIST; ++i)
      part[i] += w * c_hist[((size_t)i * BATCH + b) * DIM + d];
  }
#pragma unroll
  for (int i = 0; i < NHIST; ++i) atomicAdd(&sl[i], part[i]);
  __syncthreads();

  if (t == 0) {
    const float bw = *bwa;
    float mx = -1e30f;
    for (int i = 0; i < NHIST; ++i) {
      const float x = (i <= step) ? (sl[i] + bw) : -1e9f;
      sl[i] = x; mx = fmaxf(mx, x);
    }
    float s = 0.f;
    for (int i = 0; i < NHIST; ++i) { sl[i] = __expf(sl[i] - mx); s += sl[i]; }
    const float inv = 1.f / s;
    for (int i = 0; i < NHIST; ++i) sl[i] *= inv;
  }
  __syncthreads();

  for (int d = t; d < DIM; d += 256) {
    float acc = 0.f;
#pragma unroll
    for (int i = 0; i < NHIST; ++i)
      acc += sl[i] * m_hist[((size_t)i * BATCH + b) * DIM + d];
    catA[(size_t)b * 1024 + d] = (__bf16)acc;        // m_sa bf16 (first half of [m_sa|_m1])
  }
}

// m_next = sigmoid(glin)*m_prev + (1-sigmoid)*_m2 ; also refresh bf16 m_prev slot.
__global__ void gate_update(const float* __restrict__ glin, const float* __restrict__ m_prev,
                            const float* __restrict__ m2, float* __restrict__ m_next,
                            __bf16* __restrict__ catM)
{
  const int i = blockIdx.x * blockDim.x + threadIdx.x;
  if (i >= BATCH * DIM) return;
  const float g = 1.f / (1.f + __expf(-glin[i]));
  const float v = g * m_prev[i] + (1.f - g) * m2[i];
  m_next[i] = v;
  const int b = i >> 9, d = i & (DIM - 1);
  catM[(size_t)b * 1024 + DIM + d] = (__bf16)v;
}

// ---------------------------------------------------------------------------

extern "C" void kernel_launch(void* const* d_in, const int* in_sizes, int n_in,
                              void* d_out, int out_size, void* d_ws, size_t ws_size,
                              hipStream_t stream)
{
  (void)in_sizes; (void)n_in; (void)out_size; (void)ws_size;
  const float* q    = (const float*)d_in[0];
  const float* cws  = (const float*)d_in[1];
  const float* KB   = (const float*)d_in[2];
  const float* c0   = (const float*)d_in[3];
  const float* m0   = (const float*)d_in[4];
  const float* Wq   = (const float*)d_in[5];
  const float* bq   = (const float*)d_in[6];
  const float* Wct  = (const float*)d_in[7];
  const float* bct  = (const float*)d_in[8];
  const float* wca  = (const float*)d_in[9];
  const float* bca  = (const float*)d_in[10];
  const float* Wm   = (const float*)d_in[11];
  const float* bm   = (const float*)d_in[12];
  const float* Wkb  = (const float*)d_in[13];
  const float* bkb  = (const float*)d_in[14];
  const float* Wrm  = (const float*)d_in[15];
  const float* brm  = (const float*)d_in[16];
  const float* wra  = (const float*)d_in[17];
  const float* bra  = (const float*)d_in[18];
  const float* Wwm  = (const float*)d_in[19];
  const float* bwm  = (const float*)d_in[20];
  const float* wwa  = (const float*)d_in[21];
  const float* bwa  = (const float*)d_in[22];
  const float* Wam  = (const float*)d_in[23];
  const float* bam  = (const float*)d_in[24];
  const float* Wg   = (const float*)d_in[25];
  const float* bg   = (const float*)d_in[26];

  // ---- workspace carve-up ----
  size_t off = 0;
  auto carve = [&](size_t bytes) -> char* {
    size_t r = (off + 255) & ~(size_t)255;
    off = r + bytes;
    return (char*)d_ws + r;
  };
  const size_t BD = (size_t)BATCH * DIM;           // 65536
  __bf16* KBp_b   = (__bf16*)carve((size_t)BATCH * HWP * DIM * 2);
  __bf16* KB2_b   = (__bf16*)carve((size_t)BATCH * HWP * DIM * 2);
  __bf16* pWq     = (__bf16*)carve((size_t)DIM * DIM * 2);
  __bf16* pWct    = (__bf16*)carve((size_t)2 * DIM * DIM * 2);
  __bf16* pWm     = (__bf16*)carve((size_t)DIM * DIM * 2);
  __bf16* pWkb    = (__bf16*)carve((size_t)DIM * DIM * 2);
  __bf16* pWwm    = (__bf16*)carve((size_t)2 * DIM * DIM * 2);
  __bf16* pWam    = (__bf16*)carve((size_t)2 * DIM * DIM * 2);
  __bf16* pWg     = (__bf16*)carve((size_t)DIM * DIM * 2);
  __bf16* pV      = (__bf16*)carve((size_t)DIM * BATCH * 2);
  __bf16* Wrm_b   = (__bf16*)carve((size_t)2 * DIM * DIM * 2);
  __bf16* q_b     = (__bf16*)carve(BD * 2);
  float*  c_hist  = (float*)carve((size_t)NHIST * BD * 4);
  float*  m_hist  = (float*)carve((size_t)NHIST * BD * 4);
  __bf16* catC    = (__bf16*)carve((size_t)BATCH * 1024 * 2);   // [q_i | c_prev/c_i]
  __bf16* catM    = (__bf16*)carve((size_t)BATCH * 1024 * 2);   // [m_new | m_prev]
  __bf16* catA    = (__bf16*)carve((size_t)BATCH * 1024 * 2);   // [m_sa | _m1]
  float*  cq      = (float*)carve(BD * 4);
  float*  mp      = (float*)carve(BD * 4);
  float*  Ubuf    = (float*)carve((size_t)2 * DIM * BATCH * 4); // 1024 x 128
  float*  m2buf   = (float*)carve(BD * 4);
  float*  glin    = (float*)carve(BD * 4);

  auto ceil_div = [](int a, int b) { return (a + b - 1) / b; };

  // ---- one-time precompute ----
  zero_f32<<<ceil_div(NHIST * (int)BD, 256), 256, 0, stream>>>(c_hist, NHIST * (int)BD);
  zero_f32<<<ceil_div(NHIST * (int)BD, 256), 256, 0, stream>>>(m_hist, NHIST * (int)BD);
  copy_f32<<<ceil_div((int)BD, 256), 256, 0, stream>>>(c_hist, c0, (int)BD);
  copy_f32<<<ceil_div((int)BD, 256), 256, 0, stream>>>(m_hist, m0, (int)BD);

  cvt_bf16<<<ceil_div((int)BD, 256), 256, 0, stream>>>(q, q_b, BATCH, DIM, DIM);
  cvt_bf16<<<ceil_div((int)BD, 256), 256, 0, stream>>>(c0, catC + DIM, BATCH, DIM, 1024);
  cvt_bf16<<<ceil_div((int)BD, 256), 256, 0, stream>>>(m0, catM + DIM, BATCH, DIM, 1024);
  cvt_bf16<<<ceil_div(2 * DIM * DIM, 256), 256, 0, stream>>>(Wrm, Wrm_b, 2 * DIM, DIM, DIM);

  auto pack = [&](const float* W, __bf16* Wp, int K, int N) {
    const int total = (K >> 5) * (N >> 4) * 256;
    pack_weight<<<ceil_div(total, 256), 256, 0, stream>>>(W, Wp, K, N);
  };
  pack(Wq,  pWq,  DIM,     DIM);
  pack(Wct, pWct, 2 * DIM, DIM);
  pack(Wm,  pWm,  DIM,     DIM);
  pack(Wkb, pWkb, DIM,     DIM);
  pack(Wwm, pWwm, 2 * DIM, DIM);
  pack(Wam, pWam, 2 * DIM, DIM);
  pack(Wg,  pWg,  DIM,     DIM);

  transpose_kb<<<dim3(7, DIM / 32, BATCH), dim3(32, 8), 0, stream>>>(KB, KBp_b);

  auto gemm = [&](const __bf16* A, int lda, const __bf16* Wp, const float* bias,
                  float* Cf, int ldc, __bf16* Cb, int ldcb, int M, int N, int K) {
    gemm_bf16_wmma<<<dim3(N / 64, M / 128), 128, 0, stream>>>(A, lda, Wp, bias,
                                                              Cf, ldc, Cb, ldcb, N, K);
  };

  // KB2 = KBp @ Wkb + bkb  (M = 25088)
  gemm(KBp_b, DIM, pWkb, bkb, nullptr, 0, KB2_b, DIM, BATCH * HWP, DIM, DIM);
  // q_i = q @ Wq + bq  -> bf16 into catC[:, 0:512]
  gemm(q_b, DIM, pWq, bq, nullptr, 0, catC, 1024, BATCH, DIM, DIM);

  // ---- recurrence (T fixed at 12 per reference) ----
  for (int t = 0; t < TSTEPS; ++t) {
    float* c_cur  = c_hist + (size_t)(t + 1) * BD;
    float* m_prev = m_hist + (size_t)t * BD;
    float* m_next = m_hist + (size_t)(t + 1) * BD;

    // ControlUnit
    gemm(catC, 1024, pWct, bct, cq, DIM, nullptr, 0, BATCH, DIM, 2 * DIM);
    control_attn<<<BATCH, 256, 0, stream>>>(cq, cws, wca, bca, c_cur, catC);

    // ReadUnit (collapsed):
    gemm(catM + DIM, 1024, pWm, bm, mp, DIM, nullptr, 0, BATCH, DIM, DIM); // mp = m_prev@Wm+bm
    pack_V<<<ceil_div(32768, 256), 256, 0, stream>>>(c_cur, wra, pV);
    gemm(Wrm_b, DIM, pV, nullptr, Ubuf, BATCH, nullptr, 0, 2 * DIM, BATCH, DIM); // U = Wrm@V
    read_unit<<<BATCH, 256, 0, stream>>>(KBp_b, KB2_b, mp, Ubuf, c_cur, wra, brm, bra, catM);

    // WriteUnit
    gemm(catM, 1024, pWwm, bwm, nullptr, 0, catA + DIM, 1024, BATCH, DIM, 2 * DIM); // _m1
    write_selfattn<<<BATCH, 256, 0, stream>>>(c_cur, c_hist, m_hist, wwa, bwa, t, catA);
    gemm(catA, 1024, pWam, bam, m2buf, DIM, nullptr, 0, BATCH, DIM, 2 * DIM);       // _m2
    gemm(catC + DIM, 1024, pWg, bg, glin, DIM, nullptr, 0, BATCH, DIM, DIM);        // gate
    gate_update<<<ceil_div((int)BD, 256), 256, 0, stream>>>(glin, m_prev, m2buf, m_next, catM);
  }

  copy_f32<<<ceil_div((int)BD, 256), 256, 0, stream>>>((float*)d_out,
                                                       m_hist + (size_t)TSTEPS * BD, (int)BD);
}
